// VAGNN_81681688035581
// MI455X (gfx1250) — matrix operand order repaired and checked by reading, not compiled, not run on hardware
//
#include <hip/hip_runtime.h>
#include <hip/hip_bf16.h>

typedef long long i64;
typedef __attribute__((ext_vector_type(16))) _Float16 v16h;
typedef __attribute__((ext_vector_type(8)))  float    v8f;

#define DIM 64

// ---------------- elementwise / utility kernels ----------------
__global__ void k_zero(float* __restrict__ p, i64 n) {
    i64 i = (i64)blockIdx.x * blockDim.x + threadIdx.x;
    if (i < n) p[i] = 0.f;
}

__global__ void k_concat2(const float* __restrict__ a, i64 na,
                          const float* __restrict__ b, i64 nb,
                          float* __restrict__ cur, float* __restrict__ acc) {
    i64 i = (i64)blockIdx.x * blockDim.x + threadIdx.x;
    if (i >= na + nb) return;
    float v = (i < na) ? a[i] : b[i - na];
    cur[i] = v;
    acc[i] = v;
}

__global__ void k_add(float* __restrict__ acc, const float* __restrict__ x, i64 n) {
    i64 i = (i64)blockIdx.x * blockDim.x + threadIdx.x;
    if (i < n) acc[i] += x[i];
}

__global__ void k_scale(float* __restrict__ p, i64 n, float s) {
    i64 i = (i64)blockIdx.x * blockDim.x + threadIdx.x;
    if (i < n) p[i] *= s;
}

// COO SpMM: y[rows[e],:] += vals[e] * x[cols[e],:].  16 threads per edge, 4 floats each.
__global__ void k_spmm(const int* __restrict__ rows, const int* __restrict__ cols,
                       const float* __restrict__ vals, const float* __restrict__ x,
                       float* __restrict__ y, int nnz) {
    i64 tid = (i64)blockIdx.x * blockDim.x + threadIdx.x;
    int e = (int)(tid >> 4);
    if (e >= nnz) return;
    int dd = ((int)tid & 15) << 2;
    int r = rows[e], c = cols[e];
    float v = vals[e];
    const float4 xv = *(const float4*)(x + (i64)c * DIM + dd);
    float* yp = y + (i64)r * DIM + dd;
    atomicAdd(yp + 0, v * xv.x);
    atomicAdd(yp + 1, v * xv.y);
    atomicAdd(yp + 2, v * xv.z);
    atomicAdd(yp + 3, v * xv.w);
}

// One wave32 per row; each lane handles 2 of 64 columns.
__global__ __launch_bounds__(32) void k_l2norm(float* __restrict__ x, int nrows) {
    int n = blockIdx.x;
    int l = threadIdx.x;
    float a = x[(i64)n * DIM + l];
    float b = x[(i64)n * DIM + 32 + l];
    float s = a * a + b * b;
#pragma unroll
    for (int off = 16; off; off >>= 1) s += __shfl_xor(s, off, 32);
    float inv = 1.f / fmaxf(sqrtf(s), 1e-12f);
    x[(i64)n * DIM + l]      = a * inv;
    x[(i64)n * DIM + 32 + l] = b * inv;
}

__global__ void k_gather_f16(const float* __restrict__ src, const int* __restrict__ idx,
                             i64 nrows, _Float16* __restrict__ dst) {
    i64 i = (i64)blockIdx.x * blockDim.x + threadIdx.x;
    if (i >= nrows * DIM) return;
    i64 r = i >> 6;
    int dd = (int)i & 63;
    dst[i] = (_Float16)src[(i64)idx[r] * DIM + dd];
}

__global__ void k_conv_f16(const float* __restrict__ src, i64 n, _Float16* __restrict__ dst) {
    i64 i = (i64)blockIdx.x * blockDim.x + threadIdx.x;
    if (i < n) dst[i] = (_Float16)src[i];
}

// w[n] = sigmoid( ((atom_v[n]+non_v[n])/2 @ q) . ((atom_a[vl]+non_a[vl])/2 ) )
__global__ __launch_bounds__(64) void k_w(const float* __restrict__ atom_v,
                                          const float* __restrict__ non_v,
                                          const float* __restrict__ atom_a,
                                          const float* __restrict__ non_a,
                                          const int* __restrict__ vlist,
                                          const float* __restrict__ q,
                                          float* __restrict__ w) {
    __shared__ float vid[DIM];
    __shared__ float red[DIM];
    int n = blockIdx.x;
    int d = threadIdx.x;
    vid[d] = 0.5f * (atom_v[(i64)n * DIM + d] + non_v[(i64)n * DIM + d]);
    __syncthreads();
    float t = 0.f;
#pragma unroll 8
    for (int k = 0; k < DIM; ++k) t += vid[k] * q[k * DIM + d];
    int a = vlist[n];
    float vlog = 0.5f * (atom_a[(i64)a * DIM + d] + non_a[(i64)a * DIM + d]);
    red[d] = t * vlog;
    __syncthreads();
    for (int s = 32; s > 0; s >>= 1) {
        if (d < s) red[d] += red[d + s];
        __syncthreads();
    }
    if (d == 0) w[n] = 1.f / (1.f + __expf(-red[0]));
}

// ---------------- fused WMMA rating kernel ----------------
// A-fragment (16x32 f16): lanes 0-15 hold M=0-15 K={kb..kb+7, kb+16..kb+23},
//                         lanes 16-31 hold K={kb+8..kb+15, kb+24..kb+31}.
__device__ __forceinline__ v16h load_fragA(const _Float16* __restrict__ base, int lane, int kb) {
    int r  = lane & 15;
    int hi = (lane >> 4) << 3;                 // 0 or 8
    const _Float16* p = base + (i64)r * DIM + kb + hi;
    union { v16h v; uint4 u[2]; } f;
    f.u[0] = *(const uint4*)(p);               // K = kb+hi .. +7
    f.u[1] = *(const uint4*)(p + 16);          // K = kb+16+hi .. +7
    return f.v;
}

// B-fragment (32x16 f16), B[k][n] = rowmajor[n][k]:
// lanes 0-15 hold K={kb..kb+7, kb+16..kb+23} of column N=lane, lanes 16-31 hold K={kb+8..kb+15, kb+24..kb+31}.
__device__ __forceinline__ v16h load_fragB(const _Float16* __restrict__ base, int lane, int kb) {
    int n  = lane & 15;
    int kh = (lane >> 4) << 3;                 // 0 or 8
    const _Float16* p = base + (i64)n * DIM + kb + kh;
    union { v16h v; uint4 u[2]; } f;
    f.u[0] = *(const uint4*)(p);
    f.u[1] = *(const uint4*)(p + 16);
    return f.v;
}

__device__ __forceinline__ v8f wmma16(v16h a, v16h b, v8f c) {
    return __builtin_amdgcn_wmma_f32_16x16x32_f16(false, a, false, b, (short)0, c, false, false);
}

// One wave per 16x16 output tile; 8 WMMAs/tile (two K=64 GEMMs x two terms each).
__global__ __launch_bounds__(128) void k_rating(
    const _Float16* __restrict__ uA, const _Float16* __restrict__ uN,
    const _Float16* __restrict__ Va, const _Float16* __restrict__ Vn,
    const _Float16* __restrict__ La, const _Float16* __restrict__ Ln,
    const float* __restrict__ w, float* __restrict__ out,
    int mtiles, int Vtot) {
    int wave = threadIdx.x >> 5;
    int lane = threadIdx.x & 31;
    i64 t  = (i64)blockIdx.x * 4 + wave;
    int mt = (int)(t % mtiles);
    int nt = (int)(t / mtiles);
    i64 aoff = (i64)(mt * 16) * DIM;
    i64 boff = (i64)(nt * 16) * DIM;

    v16h a0 = load_fragA(uA + aoff, lane, 0);
    v16h a1 = load_fragA(uA + aoff, lane, 32);
    v16h c0 = load_fragA(uN + aoff, lane, 0);
    v16h c1 = load_fragA(uN + aoff, lane, 32);

    v8f acc1 = {};
    {
        v16h b0 = load_fragB(Va + boff, lane, 0);
        v16h b1 = load_fragB(Va + boff, lane, 32);
        acc1 = wmma16(a0, b0, acc1);
        acc1 = wmma16(a1, b1, acc1);
    }
    {
        v16h b0 = load_fragB(Vn + boff, lane, 0);
        v16h b1 = load_fragB(Vn + boff, lane, 32);
        acc1 = wmma16(c0, b0, acc1);
        acc1 = wmma16(c1, b1, acc1);
    }
    v8f acc2 = {};
    {
        v16h b0 = load_fragB(La + boff, lane, 0);
        v16h b1 = load_fragB(La + boff, lane, 32);
        acc2 = wmma16(a0, b0, acc2);
        acc2 = wmma16(a1, b1, acc2);
    }
    {
        v16h b0 = load_fragB(Ln + boff, lane, 0);
        v16h b1 = load_fragB(Ln + boff, lane, 32);
        acc2 = wmma16(c0, b0, acc2);
        acc2 = wmma16(c1, b1, acc2);
    }

    // C/D layout: VGPR v, lanes 0-15 -> (M=v, N=lane); lanes 16-31 -> (M=v+8, N=lane-16)
    int ncol  = nt * 16 + (lane & 15);
    float wn  = w[ncol];
    int mbase = mt * 16 + ((lane >> 4) << 3);
#pragma unroll
    for (int v = 0; v < 8; ++v) {
        float s1 = 1.f / (1.f + __expf(-acc1[v]));
        float s2 = 1.f / (1.f + __expf(-acc2[v]));
        out[(i64)(mbase + v) * Vtot + ncol] = wn * s1 + (1.f - wn) * s2;
    }
}

// ---------------- host launcher ----------------
static inline int nblk(i64 n, int t) { return (int)((n + t - 1) / t); }

extern "C" void kernel_launch(void* const* d_in, const int* in_sizes, int n_in,
                              void* d_out, int out_size, void* d_ws, size_t ws_size,
                              hipStream_t stream) {
    const int U = 100000, V = 50000, A = 5000, B = 1024;
    const i64 N1 = (i64)U + V, N2 = (i64)U + A;

    const float* emb_user    = (const float*)d_in[0];
    const float* emb_video   = (const float*)d_in[1];
    const float* emb_vlogger = (const float*)d_in[2];
    const float* q           = (const float*)d_in[3];
    const int*   ui_r = (const int*)d_in[4];
    const int*   ui_c = (const int*)d_in[5];
    const float* ui_v = (const float*)d_in[6];
    const int*   ua_r = (const int*)d_in[7];
    const int*   ua_c = (const int*)d_in[8];
    const float* ua_v = (const float*)d_in[9];
    const int*   g2_r = (const int*)d_in[10];
    const int*   g2_c = (const int*)d_in[11];
    const float* g2_v = (const float*)d_in[12];
    const int*   g4_r = (const int*)d_in[13];
    const int*   g4_c = (const int*)d_in[14];
    const float* g4_v = (const float*)d_in[15];
    const int*   vlist = (const int*)d_in[16];
    const int*   users = (const int*)d_in[17];
    float* out = (float*)d_out;

    const int nnz_ui = in_sizes[4];
    const int nnz_ua = in_sizes[7];
    const int nnz_g2 = in_sizes[10];
    const int nnz_g4 = in_sizes[13];

    // workspace layout
    char* p = (char*)d_ws;
    float* accA = (float*)p;  p += (size_t)N1 * DIM * sizeof(float);  // UI propagate acc -> atom
    float* r1   = (float*)p;  p += (size_t)N1 * DIM * sizeof(float);  // cur (later: f16 operands)
    float* r2   = (float*)p;  p += (size_t)N1 * DIM * sizeof(float);  // nxt (later: atom_a/non_v/w)
    float* accB = (float*)p;  p += (size_t)N2 * DIM * sizeof(float);  // UA propagate acc -> non

    // region1 reuse: f16 GEMM operands
    _Float16* uAh = (_Float16*)r1;
    _Float16* uNh = uAh + (i64)B * DIM;
    _Float16* Vah = uNh + (i64)B * DIM;
    _Float16* Vnh = Vah + (i64)V * DIM;
    _Float16* Lah = Vnh + (i64)V * DIM;
    _Float16* Lnh = Lah + (i64)V * DIM;
    // region2 reuse: f32 intermediates
    float* atomA = r2;                       // [A,64]  l2norm(g2 @ atom_v)
    float* nonV  = atomA + (i64)A * DIM;     // [V,64]  l2norm(g4 @ non_u)
    float* wbuf  = nonV + (i64)V * DIM;      // [V]

    const int T = 256;
    const i64 nA = N1 * DIM, nB = N2 * DIM;

    // ---- Phase A: UI LightGCN propagate (3 layers, running mean) ----
    {
        float* cur = r1;
        float* nxt = r2;
        k_concat2<<<nblk(nA, T), T, 0, stream>>>(emb_user, (i64)U * DIM, emb_video, (i64)V * DIM, cur, accA);
        for (int l = 0; l < 3; ++l) {
            k_zero<<<nblk(nA, T), T, 0, stream>>>(nxt, nA);
            k_spmm<<<nblk((i64)nnz_ui * 16, T), T, 0, stream>>>(ui_r, ui_c, ui_v, cur, nxt, nnz_ui);
            k_add<<<nblk(nA, T), T, 0, stream>>>(accA, nxt, nA);
            float* tmp = cur; cur = nxt; nxt = tmp;
        }
        k_scale<<<nblk(nA, T), T, 0, stream>>>(accA, nA, 0.25f);
    }

    // ---- Phase B: UA LightGCN propagate (uses r1/r2 as scratch, then they are free) ----
    {
        float* cur = r1;
        float* nxt = r2;
        k_concat2<<<nblk(nB, T), T, 0, stream>>>(emb_user, (i64)U * DIM, emb_vlogger, (i64)A * DIM, cur, accB);
        for (int l = 0; l < 3; ++l) {
            k_zero<<<nblk(nB, T), T, 0, stream>>>(nxt, nB);
            k_spmm<<<nblk((i64)nnz_ua * 16, T), T, 0, stream>>>(ua_r, ua_c, ua_v, cur, nxt, nnz_ua);
            k_add<<<nblk(nB, T), T, 0, stream>>>(accB, nxt, nB);
            float* tmp = cur; cur = nxt; nxt = tmp;
        }
        k_scale<<<nblk(nB, T), T, 0, stream>>>(accB, nB, 0.25f);
    }

    const float* atom_v = accA + (i64)U * DIM;   // [V,64]
    const float* non_a  = accB + (i64)U * DIM;   // [A,64]

    // ---- Phase C: atom_a = l2norm(g2 @ atom_v), non_v = l2norm(g4 @ non_u) ----
    k_zero<<<nblk((i64)A * DIM, T), T, 0, stream>>>(atomA, (i64)A * DIM);
    k_spmm<<<nblk((i64)nnz_g2 * 16, T), T, 0, stream>>>(g2_r, g2_c, g2_v, atom_v, atomA, nnz_g2);
    k_l2norm<<<A, 32, 0, stream>>>(atomA, A);

    k_zero<<<nblk((i64)V * DIM, T), T, 0, stream>>>(nonV, (i64)V * DIM);
    k_spmm<<<nblk((i64)nnz_g4 * 16, T), T, 0, stream>>>(g4_r, g4_c, g4_v, accB /*non_u*/, nonV, nnz_g4);
    k_l2norm<<<V, 32, 0, stream>>>(nonV, V);

    // ---- Phase D: per-video mixing weight w ----
    k_w<<<V, 64, 0, stream>>>(atom_v, nonV, atomA, non_a, vlist, q, wbuf);

    // ---- Phase E: gather + f32->f16 operand prep ----
    k_gather_f16<<<nblk((i64)B * DIM, T), T, 0, stream>>>(accA, users, B, uAh);
    k_gather_f16<<<nblk((i64)B * DIM, T), T, 0, stream>>>(accB, users, B, uNh);
    k_conv_f16<<<nblk((i64)V * DIM, T), T, 0, stream>>>(atom_v, (i64)V * DIM, Vah);
    k_conv_f16<<<nblk((i64)V * DIM, T), T, 0, stream>>>(nonV, (i64)V * DIM, Vnh);
    k_gather_f16<<<nblk((i64)V * DIM, T), T, 0, stream>>>(atomA, vlist, V, Lah);
    k_gather_f16<<<nblk((i64)V * DIM, T), T, 0, stream>>>(non_a, vlist, V, Lnh);

    // ---- Phase F: fused WMMA GEMMs + sigmoid + blend -> out[1024,50000] ----
    const int mtiles = B / 16;                 // 64
    const int ntiles = V / 16;                 // 3125
    const int tiles  = mtiles * ntiles;        // 200000, 4 waves/block
    k_rating<<<tiles / 4, 128, 0, stream>>>(uAh, uNh, Vah, Vnh, Lah, Lnh, wbuf, out, mtiles, V);
}
